// SSD_59940563583006
// MI455X (gfx1250) — compile-verified, matrix operand order checked by reading, not run on hardware
//
#include <hip/hip_runtime.h>
#include <cstddef>

// ---------------------------------------------------------------------------
// Types for CDNA5 WMMA (gfx1250, wave32)
// ---------------------------------------------------------------------------
typedef __attribute__((ext_vector_type(16))) __bf16 v16bf;
typedef __attribute__((ext_vector_type(8)))  float  v8f;

union BF16Frag {
    unsigned int u[8];
    v16bf        v;
};

// ---------------------------------------------------------------------------
// fp32 -> bf16 round-to-nearest-even
// ---------------------------------------------------------------------------
__device__ __forceinline__ unsigned short f32_to_bf16_rne(float f) {
    unsigned int x = __float_as_uint(f);
    if ((x & 0x7fffffffu) > 0x7f800000u)                 // NaN: quiet it
        return (unsigned short)((x >> 16) | 0x0040u);
    unsigned int r = x + 0x7fffu + ((x >> 16) & 1u);
    return (unsigned short)(r >> 16);
}

// ---------------------------------------------------------------------------
// Features: fp32 NCHW -> bf16 NHWC (channels contiguous so im2col staging is
// a single b128 load per thread). C is a power of two for every SSD level.
// ---------------------------------------------------------------------------
__global__ void nchw_to_nhwc_bf16(const float* __restrict__ src,
                                  unsigned short* __restrict__ dst,
                                  int HW, int shiftC, int total) {
    int i = blockIdx.x * blockDim.x + threadIdx.x;
    if (i >= total) return;
    const int C   = 1 << shiftC;
    const int c   = i & (C - 1);
    const int p   = i >> shiftC;           // global pixel = b*HW + pos
    const int b   = p / HW;
    const int pos = p - b * HW;
    dst[i] = f32_to_bf16_rne(src[((size_t)b * C + c) * HW + pos]);
}

// ---------------------------------------------------------------------------
// Weights: fp32 OIHW [N][C][3][3] -> bf16 seg-major [Npad][9][C], zero-padded
// in N so the GEMM B-loads need no tail guard.  K index = seg*C + c.
// ---------------------------------------------------------------------------
__global__ void repack_weights_bf16(const float* __restrict__ src,
                                    unsigned short* __restrict__ dst,
                                    int shiftC, int Nout, int total) {
    int i = blockIdx.x * blockDim.x + threadIdx.x;
    if (i >= total) return;
    const int C   = 1 << shiftC;
    const int c   = i & (C - 1);
    const int t   = i >> shiftC;
    const int seg = t % 9;
    const int n   = t / 9;
    float v = 0.f;
    if (n < Nout) v = src[((size_t)n * C + c) * 9 + seg];
    dst[i] = f32_to_bf16_rne(v);
}

// ---------------------------------------------------------------------------
// Implicit-im2col 3x3/pad1 conv as WMMA bf16 GEMM.
//   A: [M x K]  M = B*H*W, K = 9*C, K-order = (tap, channel)
//   B: [K x N]  repacked weights [Npad][9][C]
//   D: [M x N]  fp32 + bias
// Block = 256 threads = 8 waves.  Block tile: 64 M x 256 N.
// Wave w owns TWO 16-wide N-tiles (n0 = bID.y*256 + w*32) and FOUR
// M-accumulators each -> 8 WMMAs per K-step per wave, reusing 2 B fragments
// across 4 A fragments (halves LDS fragment traffic per FLOP).
// A (64 rows x 32 k) staged in LDS: per thread exactly one global_load_b128
// + one ds_store_b128 per K-step (NHWC makes K contiguous).
// Tap (ky,kx) is uniform per K-step -> scalar, no per-element division.
//
// Fragment packing per ISA 7.12.2 (16-bit A 16x32 / B 32x16):
//   lane l: row/col = l&15, half = l>>4
//   dwords j=0..3 at byte span [half*16, +16), j=4..7 at [32+half*16, +16)
//   -> two 16-byte loads per fragment.
// C/D (32-bit 16x16): lane -> column, vgpr r -> row r + 8*half.
// ---------------------------------------------------------------------------
__global__ __launch_bounds__(256) void wmma_conv3x3(
    const unsigned short* __restrict__ feat,   // bf16 NHWC
    const unsigned short* __restrict__ wgt,    // bf16 [Npad][9][C]
    const float* __restrict__ bias,            // fp32 [Nout]
    float* __restrict__ out,                   // fp32 [M][Nout]
    int C, int H, int W, int Nout, int Npad, int M, int HW)
{
    const int tid  = threadIdx.x;
    const int wave = tid >> 5;
    const int lane = tid & 31;
    const int half = lane >> 4;
    const int mr   = lane & 15;
    const int m0   = blockIdx.x * 64;
    const int n0   = blockIdx.y * 256 + wave * 32;   // first of 2 N-tiles
    const int n    = n0 + mr;
    const bool nact0 = (n0      < Npad);
    const bool nact1 = (n0 + 16 < Npad);
    const int K9     = 9 * C;
    const int csteps = C >> 5;

    // Staging role: 64 rows x 4 chunks of 8 channels (16B) each.
    const int sr = tid >> 2;          // row 0..63
    const int sq = tid & 3;           // 16-byte chunk 0..3
    const int sm = m0 + sr;
    const bool mok = (sm < M);
    int sb = 0, sy = 0, sx = 0;
    if (mok) {
        sb = sm / HW;
        const int pos = sm - sb * HW;
        sy = pos / W;
        sx = pos - sy * W;
    }

    __shared__ unsigned short As[64 * 40];     // row stride 80B (16B aligned)

    const v8f z8 = {0.f, 0.f, 0.f, 0.f, 0.f, 0.f, 0.f, 0.f};
    v8f acc0[4] = { z8, z8, z8, z8 };          // N-tile 0
    v8f acc1[4] = { z8, z8, z8, z8 };          // N-tile 1

    const unsigned short* wrow0 = wgt + (size_t)n * K9;
    const unsigned short* wrow1 = wgt + (size_t)(n + 16) * K9;

    for (int seg = 0; seg < 9; ++seg) {
        const int ky = seg / 3;                // uniform, 9x total
        const int kx = seg - ky * 3;
        const int iy = sy + ky - 1;
        const int ix = sx + kx - 1;
        const bool inb = mok && iy >= 0 && iy < H && ix >= 0 && ix < W;
        const long long poff = ((long long)sb * H + iy) * W + ix;
        const unsigned short* srcp = feat + poff * C + sq * 8;
        const int kseg = seg * C;

        for (int cc = 0; cc < csteps; ++cc) {
            const int cbase = cc << 5;
            // ---- stage A: one b128 per thread -------------------------------
            uint4 av = {0u, 0u, 0u, 0u};
            if (inb) av = *(const uint4*)(srcp + cbase);
            *(uint4*)(&As[sr * 40 + sq * 8]) = av;
            __syncthreads();

            // ---- two B fragments from L2-resident weights -------------------
            const int k0 = kseg + cbase;
            BF16Frag bf0, bf1;
            {
                uint4 l1 = {0u, 0u, 0u, 0u}, l2 = {0u, 0u, 0u, 0u};
                if (nact0) {
                    l1 = *(const uint4*)(wrow0 + k0 + half * 8);
                    l2 = *(const uint4*)(wrow0 + k0 + 16 + half * 8);
                }
                bf0.u[0] = l1.x; bf0.u[1] = l1.y; bf0.u[2] = l1.z; bf0.u[3] = l1.w;
                bf0.u[4] = l2.x; bf0.u[5] = l2.y; bf0.u[6] = l2.z; bf0.u[7] = l2.w;
            }
            {
                uint4 l1 = {0u, 0u, 0u, 0u}, l2 = {0u, 0u, 0u, 0u};
                if (nact1) {
                    l1 = *(const uint4*)(wrow1 + k0 + half * 8);
                    l2 = *(const uint4*)(wrow1 + k0 + 16 + half * 8);
                }
                bf1.u[0] = l1.x; bf1.u[1] = l1.y; bf1.u[2] = l1.z; bf1.u[3] = l1.w;
                bf1.u[4] = l2.x; bf1.u[5] = l2.y; bf1.u[6] = l2.z; bf1.u[7] = l2.w;
            }

            // ---- 4 M-tiles x 2 N-tiles = 8 WMMAs, B reused ------------------
            #pragma unroll
            for (int t = 0; t < 4; ++t) {
                const unsigned short* ap = &As[(t * 16 + mr) * 40 + half * 8];
                uint4 l1 = *(const uint4*)(ap);
                uint4 l2 = *(const uint4*)(ap + 16);
                BF16Frag af;
                af.u[0] = l1.x; af.u[1] = l1.y; af.u[2] = l1.z; af.u[3] = l1.w;
                af.u[4] = l2.x; af.u[5] = l2.y; af.u[6] = l2.z; af.u[7] = l2.w;
                acc0[t] = __builtin_amdgcn_wmma_f32_16x16x32_bf16(
                              false, af.v, false, bf0.v, (short)0, acc0[t],
                              false, false);
                acc1[t] = __builtin_amdgcn_wmma_f32_16x16x32_bf16(
                              false, af.v, false, bf1.v, (short)0, acc1[t],
                              false, false);
            }
            __syncthreads();
        }
    }

    if (n < Nout) {
        const float bb = bias[n];
        #pragma unroll
        for (int t = 0; t < 4; ++t) {
            #pragma unroll
            for (int r = 0; r < 8; ++r) {
                const int m = m0 + t * 16 + r + half * 8;
                if (m < M) out[(size_t)m * Nout + n] = acc0[t][r] + bb;
            }
        }
    }
    if (n + 16 < Nout) {
        const float bb = bias[n + 16];
        #pragma unroll
        for (int t = 0; t < 4; ++t) {
            #pragma unroll
            for (int r = 0; r < 8; ++r) {
                const int m = m0 + t * 16 + r + half * 8;
                if (m < M) out[(size_t)m * Nout + (n + 16)] = acc1[t][r] + bb;
            }
        }
    }
}

// ---------------------------------------------------------------------------
// Softmax over 81 classes, drop background, threshold at 0.01, scatter to
// per-image flat score array scores[b][anchor*80 + (cls-1)].
// ---------------------------------------------------------------------------
#define NSC 698560     // 8732 * 80
#define NANCH 8732

__global__ void softmax_score(const float* __restrict__ clsout,
                              float* __restrict__ scores,
                              int an, int HW, int abase, int total)
{
    int t = blockIdx.x * blockDim.x + threadIdx.x;
    if (t >= total) return;
    const int Nout = an * 81;
    const int m = t / an;
    const int a = t - m * an;
    const float* v = clsout + (size_t)m * Nout + a * 81;
    float mx = v[0];
    #pragma unroll 8
    for (int c = 1; c < 81; ++c) mx = fmaxf(mx, v[c]);
    float sum = 0.f;
    #pragma unroll 8
    for (int c = 0; c < 81; ++c) sum += __expf(v[c] - mx);
    const float inv = 1.0f / sum;
    const int b   = m / HW;
    const int pos = m - b * HW;
    const int anc = abase + pos * an + a;
    float* dst = scores + (size_t)b * NSC + (size_t)anc * 80;
    #pragma unroll 8
    for (int c = 1; c < 81; ++c) {
        float p = __expf(v[c] - mx) * inv;
        dst[c - 1] = (p > 0.01f) ? p : 0.f;
    }
}

// ---------------------------------------------------------------------------
// Exact per-image top-400 (descending, ties -> lowest index, matching
// lax.top_k). One 1024-thread block per image. Chunk-max cache + rescan.
// ---------------------------------------------------------------------------
#define NCHUNK 683       // ceil(698560 / 1024)
#define KCAND 400

__global__ __launch_bounds__(1024) void topk_select(
    float* __restrict__ scores, int* __restrict__ cand_idx,
    float* __restrict__ cand_val)
{
    const int b   = blockIdx.x;
    const int tid = threadIdx.x;
    float* sc = scores + (size_t)b * NSC;

    __shared__ float vmax[NCHUNK];
    __shared__ int   imax[NCHUNK];
    __shared__ float rv[1024];
    __shared__ int   ri[1024];
    __shared__ int   sel_c, sel_g;

    if (tid < NCHUNK) {
        const int base = tid * 1024;
        const int end  = (base + 1024 < NSC) ? base + 1024 : NSC;
        float bv = -1.0f; int bi = base;
        for (int i = base; i < end; ++i) {
            float v = sc[i];
            if (v > bv) { bv = v; bi = i; }
        }
        vmax[tid] = bv; imax[tid] = bi;
    }
    __syncthreads();

    for (int it = 0; it < KCAND; ++it) {
        rv[tid] = (tid < NCHUNK) ? vmax[tid] : -3.0f;
        ri[tid] = tid;
        __syncthreads();
        for (int s = 512; s > 0; s >>= 1) {
            if (tid < s) {
                float v2 = rv[tid + s]; int i2 = ri[tid + s];
                if (v2 > rv[tid] || (v2 == rv[tid] && i2 < ri[tid])) {
                    rv[tid] = v2; ri[tid] = i2;
                }
            }
            __syncthreads();
        }
        if (tid == 0) {
            const int   c = ri[0];
            const int   g = imax[c];
            const float v = rv[0];
            sel_c = c; sel_g = g;
            cand_idx[b * KCAND + it] = g;
            cand_val[b * KCAND + it] = (v > 0.f) ? v : 0.f;
            sc[g] = -1.0f;                       // permanently exclude
        }
        __threadfence();
        __syncthreads();

        const int c    = sel_c;
        const int gsel = sel_g;
        const int base = c * 1024;
        const int idx  = base + tid;
        float v = -1.0f;
        if (idx < NSC && idx != gsel) v = sc[idx];
        rv[tid] = v;
        ri[tid] = (idx < NSC) ? idx : base;
        __syncthreads();
        for (int s = 512; s > 0; s >>= 1) {
            if (tid < s) {
                float v2 = rv[tid + s]; int i2 = ri[tid + s];
                if (v2 > rv[tid] || (v2 == rv[tid] && i2 < ri[tid])) {
                    rv[tid] = v2; ri[tid] = i2;
                }
            }
            __syncthreads();
        }
        if (tid == 0) { vmax[c] = rv[0]; imax[c] = ri[0]; }
        __syncthreads();
    }
}

// ---------------------------------------------------------------------------
// Decode + class-aware greedy NMS + final top-200, one block per image.
// ---------------------------------------------------------------------------
#define KOUT 200
#define NMSW 13          // ceil(400/32) suppression-mask words per row

__global__ __launch_bounds__(512) void nms_kernel(
    const int* __restrict__ cand_idx, const float* __restrict__ cand_val,
    const float* __restrict__ r0, const float* __restrict__ r1,
    const float* __restrict__ r2, const float* __restrict__ r3,
    const float* __restrict__ r4, const float* __restrict__ r5,
    const float* __restrict__ priors,
    float* __restrict__ boxes_out, float* __restrict__ scores_out,
    int* __restrict__ classes_out)
{
    const int b   = blockIdx.x;
    const int tid = threadIdx.x;

    __shared__ float ox1[KCAND], oy1[KCAND], ox2[KCAND], oy2[KCAND];
    __shared__ float rb[KCAND * 4];
    __shared__ float sv[KCAND];
    __shared__ int   scls[KCAND];
    __shared__ unsigned int smask[KCAND * NMSW];
    __shared__ unsigned int keepw[NMSW];
    __shared__ int   ord[KCAND];
    __shared__ int   cnt;

    for (int i = tid; i < KCAND * NMSW; i += 512) smask[i] = 0u;
    if (tid < NMSW) keepw[tid] = 0u;
    __syncthreads();

    const float* regs[6] = { r0, r1, r2, r3, r4, r5 };

    for (int i = tid; i < KCAND; i += 512) {
        const int   g = cand_idx[b * KCAND + i];
        const float s = cand_val[b * KCAND + i];
        const int cls = g % 80 + 1;
        const int anc = g / 80;

        const int LB[7]  = {0, 5776, 7942, 8542, 8692, 8728, 8732};
        const int AN[6]  = {4, 6, 6, 6, 4, 4};
        const int HWl[6] = {1444, 361, 100, 25, 9, 1};
        int lvl = 0;
        while (lvl < 5 && anc >= LB[lvl + 1]) ++lvl;
        const int pa  = anc - LB[lvl];
        const int an  = AN[lvl];
        const int pos = pa / an;
        const int a   = pa - pos * an;
        const int m   = b * HWl[lvl] + pos;
        const float* rg = regs[lvl] + (size_t)m * (an * 4) + a * 4;

        const float* p = priors + (size_t)anc * 4;
        const float CLIP = 4.135166556742356f;     // log(1000/16)
        const float cx = p[0] + rg[0] * 0.1f * p[2];
        const float cy = p[1] + rg[1] * 0.1f * p[3];
        const float w  = p[2] * __expf(fminf(rg[2] * 0.2f, CLIP));
        const float h  = p[3] * __expf(fminf(rg[3] * 0.2f, CLIP));
        const float x1 = fminf(fmaxf(cx - 0.5f * w, 0.f), 300.f);
        const float y1 = fminf(fmaxf(cy - 0.5f * h, 0.f), 300.f);
        const float x2 = fminf(fmaxf(cx + 0.5f * w, 0.f), 300.f);
        const float y2 = fminf(fmaxf(cy + 0.5f * h, 0.f), 300.f);

        const bool valid = (s > 0.f) && (x2 > x1) && (y2 > y1);
        const float off = (float)cls * 301.0f;     // batched-NMS class offset
        ox1[i] = x1 + off; oy1[i] = y1 + off;
        ox2[i] = x2 + off; oy2[i] = y2 + off;
        rb[i * 4 + 0] = x1; rb[i * 4 + 1] = y1;
        rb[i * 4 + 2] = x2; rb[i * 4 + 3] = y2;
        sv[i]   = valid ? s : 0.f;
        scls[i] = cls;
        if (valid) atomicOr(&keepw[i >> 5], 1u << (i & 31));
    }
    __syncthreads();

    for (int pp = tid; pp < KCAND * KCAND; pp += 512) {
        const int i = pp / KCAND;
        const int j = pp - i * KCAND;
        if (j > i) {
            const float ai = (ox2[i] - ox1[i]) * (oy2[i] - oy1[i]);
            const float aj = (ox2[j] - ox1[j]) * (oy2[j] - oy1[j]);
            const float iw = fminf(ox2[i], ox2[j]) - fmaxf(ox1[i], ox1[j]);
            const float ih = fminf(oy2[i], oy2[j]) - fmaxf(oy1[i], oy1[j]);
            const float inter = fmaxf(iw, 0.f) * fmaxf(ih, 0.f);
            const float iou = inter / (ai + aj - inter + 1e-9f);
            if (iou > 0.45f)
                atomicOr(&smask[i * NMSW + (j >> 5)], 1u << (j & 31));
        }
    }
    __syncthreads();

    for (int i = 0; i < KCAND; ++i) {
        const bool kept = (keepw[i >> 5] >> (i & 31)) & 1u;
        if (kept && tid < NMSW) keepw[tid] &= ~smask[i * NMSW + tid];
        __syncthreads();
    }

    if (tid == 0) {
        int c = 0;
        for (int i = 0; i < KCAND; ++i)
            if ((keepw[i >> 5] >> (i & 31)) & 1u) ord[c++] = i;
        cnt = c;
        for (int i = 0; i < KCAND; ++i)
            if (!((keepw[i >> 5] >> (i & 31)) & 1u)) ord[c++] = i;
    }
    __syncthreads();

    for (int t = tid; t < KOUT; t += 512) {
        const int   i = ord[t];
        const float s = (t < cnt) ? sv[i] : 0.f;
        boxes_out[((size_t)b * KOUT + t) * 4 + 0] = rb[i * 4 + 0];
        boxes_out[((size_t)b * KOUT + t) * 4 + 1] = rb[i * 4 + 1];
        boxes_out[((size_t)b * KOUT + t) * 4 + 2] = rb[i * 4 + 2];
        boxes_out[((size_t)b * KOUT + t) * 4 + 3] = rb[i * 4 + 3];
        scores_out[(size_t)b * KOUT + t]  = s;
        classes_out[(size_t)b * KOUT + t] = (s > 0.f) ? scls[i] : 0;
    }
}

// ---------------------------------------------------------------------------
// Host-side orchestration (graph-capture safe: launches + pointer math only)
// ---------------------------------------------------------------------------
extern "C" void kernel_launch(void* const* d_in, const int* in_sizes, int n_in,
                              void* d_out, int out_size, void* d_ws, size_t ws_size,
                              hipStream_t stream) {
    (void)in_sizes; (void)n_in; (void)out_size; (void)ws_size;

    static const int Cs[6]  = {512, 1024, 512, 256, 256, 256};
    static const int SC[6]  = {9, 10, 9, 8, 8, 8};          // log2(C)
    static const int Hs[6]  = {38, 19, 10, 5, 3, 1};
    static const int ANs[6] = {4, 6, 6, 6, 4, 4};
    static const int ABASE[6] = {0, 5776, 7942, 8542, 8692, 8728};
    const int Bn = 16;

    int HW[6], M[6], Ncls[6], Nreg[6], Pcls[6], Preg[6];
    for (int i = 0; i < 6; ++i) {
        HW[i]   = Hs[i] * Hs[i];
        M[i]    = Bn * HW[i];
        Ncls[i] = ANs[i] * 81;
        Nreg[i] = ANs[i] * 4;
        Pcls[i] = (Ncls[i] + 15) & ~15;
        Preg[i] = (Nreg[i] + 15) & ~15;
    }

    // Workspace layout (256B aligned regions)
    char* wsb = (char*)d_ws;
    size_t cur = 0;
    auto alloc = [&](size_t bytes) -> size_t {
        size_t o = (cur + 255) & ~(size_t)255;
        cur = o + bytes;
        return o;
    };
    unsigned short *featbf[6], *clswbf[6], *regwbf[6];
    float *clsout[6], *regout[6];
    for (int i = 0; i < 6; ++i)
        featbf[i] = (unsigned short*)(wsb + alloc((size_t)Bn * Cs[i] * HW[i] * 2));
    for (int i = 0; i < 6; ++i)
        clswbf[i] = (unsigned short*)(wsb + alloc((size_t)Pcls[i] * 9 * Cs[i] * 2));
    for (int i = 0; i < 6; ++i)
        regwbf[i] = (unsigned short*)(wsb + alloc((size_t)Preg[i] * 9 * Cs[i] * 2));
    for (int i = 0; i < 6; ++i)
        clsout[i] = (float*)(wsb + alloc((size_t)M[i] * Ncls[i] * 4));
    for (int i = 0; i < 6; ++i)
        regout[i] = (float*)(wsb + alloc((size_t)M[i] * Nreg[i] * 4));
    float* scores   = (float*)(wsb + alloc((size_t)Bn * NSC * 4));
    int*   cand_idx = (int*)  (wsb + alloc((size_t)Bn * KCAND * 4));
    float* cand_val = (float*)(wsb + alloc((size_t)Bn * KCAND * 4));

    // Inputs in setup_inputs() dict order:
    // per level {feat, cls_w, cls_b, reg_w, reg_b}, then priors (index 30).
    for (int i = 0; i < 6; ++i) {
        {   // features NCHW -> NHWC bf16
            int total = Bn * HW[i] * Cs[i];
            nchw_to_nhwc_bf16<<<(total + 255) / 256, 256, 0, stream>>>(
                (const float*)d_in[i * 5 + 0], featbf[i], HW[i], SC[i], total);
        }
        {   // cls weights -> seg-major bf16, N zero-padded
            int total = Pcls[i] * 9 * Cs[i];
            repack_weights_bf16<<<(total + 255) / 256, 256, 0, stream>>>(
                (const float*)d_in[i * 5 + 1], clswbf[i], SC[i], Ncls[i], total);
        }
        {   // reg weights
            int total = Preg[i] * 9 * Cs[i];
            repack_weights_bf16<<<(total + 255) / 256, 256, 0, stream>>>(
                (const float*)d_in[i * 5 + 3], regwbf[i], SC[i], Nreg[i], total);
        }
    }
    const float* priors = (const float*)d_in[30];

    // WMMA conv heads: 64M x 256N block tiles (8 waves x 2 N-tiles each)
    for (int i = 0; i < 6; ++i) {
        {
            dim3 grid((M[i] + 63) / 64, ((Pcls[i] / 16) + 15) / 16);
            wmma_conv3x3<<<grid, 256, 0, stream>>>(
                featbf[i], clswbf[i], (const float*)d_in[i * 5 + 2], clsout[i],
                Cs[i], Hs[i], Hs[i], Ncls[i], Pcls[i], M[i], HW[i]);
        }
        {
            dim3 grid((M[i] + 63) / 64, ((Preg[i] / 16) + 15) / 16);
            wmma_conv3x3<<<grid, 256, 0, stream>>>(
                featbf[i], regwbf[i], (const float*)d_in[i * 5 + 4], regout[i],
                Cs[i], Hs[i], Hs[i], Nreg[i], Preg[i], M[i], HW[i]);
        }
    }

    // Softmax + threshold -> flat scores
    for (int i = 0; i < 6; ++i) {
        int total = M[i] * ANs[i];
        softmax_score<<<(total + 255) / 256, 256, 0, stream>>>(
            clsout[i], scores, ANs[i], HW[i], ABASE[i], total);
    }

    // Exact top-400 per image
    topk_select<<<Bn, 1024, 0, stream>>>(scores, cand_idx, cand_val);

    // Decode + NMS + top-200
    float* boxes_out   = (float*)d_out;
    float* scores_out  = boxes_out + (size_t)Bn * KOUT * 4;
    int*   classes_out = (int*)(scores_out + (size_t)Bn * KOUT);
    nms_kernel<<<Bn, 512, 0, stream>>>(cand_idx, cand_val,
                                       regout[0], regout[1], regout[2],
                                       regout[3], regout[4], regout[5],
                                       priors, boxes_out, scores_out, classes_out);
}
